// Int8MixtralBlockSparseTop2MLP_75368086110385
// MI455X (gfx1250) — 8-bit matrix operand order repaired and checked by reading, compiled, not run on hardware
//
#include <hip/hip_runtime.h>
#include <stdint.h>

typedef int v8i __attribute__((ext_vector_type(8)));
typedef int vi4 __attribute__((vector_size(16)));

#define HIDDEN 4096
#define FFN    14336
#define TOKENS 8192

#define BM 128
#define BN 128
#define BK 64
#define NTHREADS 256
#define TILE_BYTES (BM * BK)

#if __has_builtin(__builtin_amdgcn_global_load_async_to_lds_b128)
#define HAVE_ASYNC_LDS 1
#else
#define HAVE_ASYNC_LDS 0
#endif

__device__ __forceinline__ void wait_async_zero() {
#if HAVE_ASYNC_LDS
#if __has_builtin(__builtin_amdgcn_s_wait_asynccnt)
    __builtin_amdgcn_s_wait_asynccnt(0);
#else
    asm volatile("s_wait_asynccnt 0x0" ::: "memory");
#endif
#endif
}

__device__ __forceinline__ void copy16(int8_t* dst_lds, const int8_t* src_glb) {
#if HAVE_ASYNC_LDS
    __builtin_amdgcn_global_load_async_to_lds_b128(
        (__attribute__((address_space(1))) vi4*)src_glb,
        (__attribute__((address_space(3))) vi4*)dst_lds, 0, 0);
#else
    *(uint4*)dst_lds = *(const uint4*)src_glb;
#endif
}

__device__ __forceinline__ void issue_tile(int8_t* __restrict__ dst,
                                           const int8_t* __restrict__ src,
                                           long row0, long col0, long ld, int tid) {
    const int c0 = tid;
    const int c1 = tid + NTHREADS;
    copy16(dst + (c0 >> 2) * BK + (c0 & 3) * 16,
           src + (row0 + (c0 >> 2)) * ld + col0 + (c0 & 3) * 16);
    copy16(dst + (c1 >> 2) * BK + (c1 & 3) * 16,
           src + (row0 + (c1 >> 2)) * ld + col0 + (c1 & 3) * 16);
}

__device__ __forceinline__ v8i load_a_frag(const int8_t* rowbase, int khalf) {
    const int off[8] = {0, 4, 16, 20, 32, 36, 48, 52};
    v8i f;
#pragma unroll
    for (int j = 0; j < 8; ++j)
        f[j] = *(const int*)(rowbase + off[j] + khalf);
    return f;
}

__device__ __forceinline__ v8i load_b_frag(const int8_t* rowbase, int khalf) {
    const int off[8] = {0, 4, 16, 20, 32, 36, 48, 52};
    v8i f;
#pragma unroll
    for (int j = 0; j < 8; ++j)
        f[j] = *(const int*)(rowbase + off[j] + khalf);
    return f;
}

__global__ __launch_bounds__(256) void quantize_x(const float* __restrict__ x,
                                                  int8_t* __restrict__ qx,
                                                  const float* __restrict__ a1_s) {
    const long i = (long)blockIdx.x * blockDim.x + threadIdx.x;
    const float inv = 1.0f / a1_s[0];
    const float4 v = ((const float4*)x)[i];
    int b0 = (int)fminf(127.f, fmaxf(-128.f, rintf(v.x * inv)));
    int b1 = (int)fminf(127.f, fmaxf(-128.f, rintf(v.y * inv)));
    int b2 = (int)fminf(127.f, fmaxf(-128.f, rintf(v.z * inv)));
    int b3 = (int)fminf(127.f, fmaxf(-128.f, rintf(v.w * inv)));
    const int packed = (b0 & 0xFF) | ((b1 & 0xFF) << 8) |
                       ((b2 & 0xFF) << 16) | ((b3 & 0xFF) << 24);
    ((int*)qx)[i] = packed;
}

__global__ __launch_bounds__(NTHREADS) void gemm_gate_up(
    const int8_t* __restrict__ qx, const int8_t* __restrict__ w1,
    const int8_t* __restrict__ w3, int8_t* __restrict__ qh,
    const float* __restrict__ w1_s, const float* __restrict__ w3_s,
    const float* __restrict__ a1_s, const float* __restrict__ a2_s) {
    __shared__ __align__(16) int8_t smem[6 * TILE_BYTES];
    int8_t* As  = smem;
    int8_t* B1s = smem + 2 * TILE_BYTES;
    int8_t* B3s = smem + 4 * TILE_BYTES;

    const int tid  = threadIdx.x;
    const int lane = tid & 31;
    const int wid  = tid >> 5;
    const int wm   = (wid >> 1) * 32;
    const int wn   = (wid & 1) * 64;
    const long m0  = (long)blockIdx.y * BM;
    const long n0  = (long)blockIdx.x * BN;

    const int lrow = lane & 15;
    const int khA  = (lane < 16) ? 0 : 8;
    const int khB  = (lane < 16) ? 0 : 8;

    const v8i vzero = {0, 0, 0, 0, 0, 0, 0, 0};
    v8i accg[2][4], accu[2][4];
#pragma unroll
    for (int mt = 0; mt < 2; ++mt)
#pragma unroll
        for (int nt = 0; nt < 4; ++nt) { accg[mt][nt] = vzero; accu[mt][nt] = vzero; }

    issue_tile(As,  qx, m0, 0, HIDDEN, tid);
    issue_tile(B1s, w1, n0, 0, HIDDEN, tid);
    issue_tile(B3s, w3, n0, 0, HIDDEN, tid);
    wait_async_zero();
    __syncthreads();

    int buf = 0;
    for (int kk = 0; kk < HIDDEN; kk += BK) {
        const int nxt = buf ^ 1;
        if (kk + BK < HIDDEN) {
            issue_tile(As  + nxt * TILE_BYTES, qx, m0, kk + BK, HIDDEN, tid);
            issue_tile(B1s + nxt * TILE_BYTES, w1, n0, kk + BK, HIDDEN, tid);
            issue_tile(B3s + nxt * TILE_BYTES, w3, n0, kk + BK, HIDDEN, tid);
        }

        const int8_t* Ab  = As  + buf * TILE_BYTES;
        const int8_t* B1b = B1s + buf * TILE_BYTES;
        const int8_t* B3b = B3s + buf * TILE_BYTES;

        v8i a[2];
        a[0] = load_a_frag(Ab + (wm + lrow) * BK, khA);
        a[1] = load_a_frag(Ab + (wm + 16 + lrow) * BK, khA);

#pragma unroll
        for (int nt = 0; nt < 4; ++nt) {
            const v8i b1 = load_b_frag(B1b + (wn + nt * 16 + lrow) * BK, khB);
            const v8i b3 = load_b_frag(B3b + (wn + nt * 16 + lrow) * BK, khB);
#pragma unroll
            for (int mt = 0; mt < 2; ++mt) {
                accg[mt][nt] = __builtin_amdgcn_wmma_i32_16x16x64_iu8(
                    true, a[mt], true, b1, accg[mt][nt], false, false);
                accu[mt][nt] = __builtin_amdgcn_wmma_i32_16x16x64_iu8(
                    true, a[mt], true, b3, accu[mt][nt], false, false);
            }
        }

        wait_async_zero();
        __syncthreads();
        buf = nxt;
    }

    const float sg     = a1_s[0] * w1_s[0];
    const float su     = a1_s[0] * w3_s[0];
    const float inv_a2 = 1.0f / a2_s[0];
    const int   ncol   = lrow;
    const int   mbase  = (lane < 16) ? 0 : 8;
    int8_t* ep = smem;
#pragma unroll
    for (int mt = 0; mt < 2; ++mt)
#pragma unroll
        for (int nt = 0; nt < 4; ++nt)
#pragma unroll
            for (int r = 0; r < 8; ++r) {
                const int lr = wm + mt * 16 + mbase + r;
                const int lc = wn + nt * 16 + ncol;
                const float g = (float)accg[mt][nt][r] * sg;
                const float u = (float)accu[mt][nt][r] * su;
                const float h = (g / (1.0f + __expf(-g))) * u;
                float qf = rintf(h * inv_a2);
                qf = fminf(127.f, fmaxf(-128.f, qf));
                ep[lr * BN + lc] = (int8_t)(int)qf;
            }
    __syncthreads();
#pragma unroll
    for (int j = 0; j < 4; ++j) {
        const int c   = tid + j * NTHREADS;
        const int row = c >> 3;
        const int col = (c & 7) * 16;
        *(uint4*)(qh + (m0 + row) * FFN + n0 + col) =
            *(const uint4*)(ep + row * BN + col);
    }
}

__global__ __launch_bounds__(NTHREADS) void gemm_down(
    const int8_t* __restrict__ qh, const int8_t* __restrict__ w2,
    float* __restrict__ out,
    const float* __restrict__ w2_s, const float* __restrict__ a2_s) {
    __shared__ __align__(16) int8_t smem[4 * TILE_BYTES];
    int8_t* As = smem;
    int8_t* Bs = smem + 2 * TILE_BYTES;

    const int tid  = threadIdx.x;
    const int lane = tid & 31;
    const int wid  = tid >> 5;
    const int wm   = (wid >> 1) * 32;
    const int wn   = (wid & 1) * 64;
    const long m0  = (long)blockIdx.y * BM;
    const long n0  = (long)blockIdx.x * BN;

    const int lrow = lane & 15;
    const int khA  = (lane < 16) ? 0 : 8;
    const int khB  = (lane < 16) ? 0 : 8;

    const v8i vzero = {0, 0, 0, 0, 0, 0, 0, 0};
    v8i acc[2][4];
#pragma unroll
    for (int mt = 0; mt < 2; ++mt)
#pragma unroll
        for (int nt = 0; nt < 4; ++nt) acc[mt][nt] = vzero;

    issue_tile(As, qh, m0, 0, FFN, tid);
    issue_tile(Bs, w2, n0, 0, FFN, tid);
    wait_async_zero();
    __syncthreads();

    int buf = 0;
    for (int kk = 0; kk < FFN; kk += BK) {
        const int nxt = buf ^ 1;
        if (kk + BK < FFN) {
            issue_tile(As + nxt * TILE_BYTES, qh, m0, kk + BK, FFN, tid);
            issue_tile(Bs + nxt * TILE_BYTES, w2, n0, kk + BK, FFN, tid);
        }

        const int8_t* Ab = As + buf * TILE_BYTES;
        const int8_t* Bb = Bs + buf * TILE_BYTES;

        v8i a[2];
        a[0] = load_a_frag(Ab + (wm + lrow) * BK, khA);
        a[1] = load_a_frag(Ab + (wm + 16 + lrow) * BK, khA);

#pragma unroll
        for (int nt = 0; nt < 4; ++nt) {
            const v8i b = load_b_frag(Bb + (wn + nt * 16 + lrow) * BK, khB);
#pragma unroll
            for (int mt = 0; mt < 2; ++mt) {
                acc[mt][nt] = __builtin_amdgcn_wmma_i32_16x16x64_iu8(
                    true, a[mt], true, b, acc[mt][nt], false, false);
            }
        }

        wait_async_zero();
        __syncthreads();
        buf = nxt;
    }

    const float so    = a2_s[0] * w2_s[0];
    const int   ncol  = lrow;
    const int   mbase = (lane < 16) ? 0 : 8;
    float* ep = (float*)smem;
    const int half = (wm >= 64) ? 1 : 0;
#pragma unroll
    for (int p = 0; p < 2; ++p) {
        if (p) __syncthreads();
        if (half == p) {
#pragma unroll
            for (int mt = 0; mt < 2; ++mt)
#pragma unroll
                for (int nt = 0; nt < 4; ++nt)
#pragma unroll
                    for (int r = 0; r < 8; ++r) {
                        const int lr = wm - p * 64 + mt * 16 + mbase + r;
                        const int lc = wn + nt * 16 + ncol;
                        ep[lr * BN + lc] = (float)acc[mt][nt][r] * so;
                    }
        }
        __syncthreads();
#pragma unroll
        for (int j = 0; j < 8; ++j) {
            const int c   = tid + j * NTHREADS;
            const int row = c >> 5;
            const int col = (c & 31) * 4;
            *(float4*)(out + (m0 + p * 64 + row) * HIDDEN + n0 + col) =
                *(const float4*)(ep + row * BN + col);
        }
    }
}

extern "C" void kernel_launch(void* const* d_in, const int* in_sizes, int n_in,
                              void* d_out, int out_size, void* d_ws, size_t ws_size,
                              hipStream_t stream) {
    const float*  x    = (const float*)d_in[0];
    const int8_t* w1   = (const int8_t*)d_in[1];
    const int8_t* w3   = (const int8_t*)d_in[2];
    const int8_t* w2   = (const int8_t*)d_in[3];
    const float*  w1_s = (const float*)d_in[4];
    const float*  w3_s = (const float*)d_in[5];
    const float*  w2_s = (const float*)d_in[6];
    const float*  a1_s = (const float*)d_in[7];
    const float*  a2_s = (const float*)d_in[8];
    float* out = (float*)d_out;

    int8_t* qx = (int8_t*)d_ws;
    int8_t* qh = (int8_t*)d_ws + (size_t)TOKENS * HIDDEN;

    {
        const long n4 = (long)TOKENS * HIDDEN / 4;
        quantize_x<<<(int)(n4 / 256), 256, 0, stream>>>(x, qx, a1_s);
    }
    {
        dim3 grid(FFN / BN, TOKENS / BM);
        gemm_gate_up<<<grid, NTHREADS, 0, stream>>>(qx, w1, w3, qh,
                                                    w1_s, w3_s, a1_s, a2_s);
    }
    {
        dim3 grid(HIDDEN / BN, TOKENS / BM);
        gemm_down<<<grid, NTHREADS, 0, stream>>>(qh, w2, out, w2_s, a2_s);
    }
}
